// TripleLoss_61710090109327
// MI455X (gfx1250) — matrix operand order repaired and checked by reading, not compiled, not run on hardware
//
#include <hip/hip_runtime.h>
#include <hip/hip_bf16.h>

typedef __attribute__((ext_vector_type(16))) _Float16 v16h;
typedef __attribute__((ext_vector_type(8)))  _Float16 v8h;
typedef __attribute__((ext_vector_type(8)))  float    v8f;

#define NN 8192
#define DD 256
#define NTILES (NN / 16)   // 512 column tiles
#define MROWS 32           // rows per block (two 16-row WMMA tiles per wave)

// ---------------- Phase 1: row L2-normalize fp32 -> f16 ----------------
__global__ __launch_bounds__(256) void k_normalize(const float* __restrict__ x,
                                                   _Float16* __restrict__ xn) {
  const int row = blockIdx.x;
  const int tid = threadIdx.x;
  float v = x[(size_t)row * DD + tid];
  float ss = v * v;
  #pragma unroll
  for (int m = 16; m >= 1; m >>= 1) ss += __shfl_xor(ss, m, 32);
  __shared__ float sp[8];
  __shared__ float snorm;
  if ((tid & 31) == 0) sp[tid >> 5] = ss;
  __syncthreads();
  if (tid == 0) {
    float t = 0.f;
    #pragma unroll
    for (int i = 0; i < 8; ++i) t += sp[i];
    snorm = sqrtf(t) + 1e-12f;
  }
  __syncthreads();
  xn[(size_t)row * DD + tid] = (_Float16)(v / snorm);
}

// ------- Phase 2: fused Gram-matrix WMMA + hard mining + row loss -------
// Block owns 32 rows (2 M-tiles per wave, B fragment reused across both).
// 8 waves split the 512 column tiles.
__global__ __launch_bounds__(256) void k_mine(const float* __restrict__ x,
                                              const _Float16* __restrict__ xn,
                                              const int* __restrict__ labels,
                                              float* __restrict__ rowloss) {
  const int mBase = blockIdx.x * MROWS;   // rows [mBase, mBase+32)
  const int tid   = threadIdx.x;
  const int wave  = tid >> 5;             // 8 waves per block (wave32)
  const int lane  = tid & 31;
  const int h     = lane >> 4;            // lane half (0/1)
  const int l16   = lane & 15;

  // Preload A fragments for both 16-row tiles (K = 256 total, 8 k-blocks).
  // A layout (16-bit 16x32): lanes 0-15 row=l16, K=kb*32+0..15;
  //                          lanes 16-31 same rows, K shifted by +16.
  v16h afrag[2][8];
  #pragma unroll
  for (int t = 0; t < 2; ++t) {
    const _Float16* arow = xn + (size_t)(mBase + t * 16 + l16) * DD + h * 16;
    #pragma unroll
    for (int kb = 0; kb < 8; ++kb) {
      union { v16h v; v8h p[2]; } u;
      u.p[0] = *(const v8h*)(arow + kb * 32);
      u.p[1] = *(const v8h*)(arow + kb * 32 + 8);
      afrag[t][kb] = u.v;
    }
  }

  // Row labels: C vgpr v of tile t holds row (t*16 + v + 8*h).
  int lr[2][8];
  #pragma unroll
  for (int t = 0; t < 2; ++t)
    #pragma unroll
    for (int v = 0; v < 8; ++v) lr[t][v] = labels[mBase + t * 16 + 8 * h + v];

  float mn[2][8], mx[2][8];
  #pragma unroll
  for (int t = 0; t < 2; ++t)
    #pragma unroll
    for (int v = 0; v < 8; ++v) { mn[t][v] = __builtin_inff(); mx[t][v] = -__builtin_inff(); }

  // Sweep column tiles; wave w handles tiles w, w+8, ... (64 each, uniform).
  for (int ct = wave; ct < NTILES; ct += 8) {
    const int cBase = ct * 16;
    const int lc = labels[cBase + l16];           // this lane's column label
    // B layout (32x16 f16): lane col n=l16, K = kb*32 + 16*h + e (contiguous
    // halves of a row of xn -- Gram symmetry, no transpose needed).
    const _Float16* bp = xn + (size_t)(cBase + l16) * DD + h * 16;
    // Prefetch this lane's B row for the next tile this wave will touch.
    {
      const int ctn = (ct + 8 < NTILES) ? (ct + 8) : ct;
      __builtin_prefetch(xn + (size_t)(ctn * 16 + l16) * DD + h * 16, 0, 1);
    }
    v8f c0 = {}, c1 = {};
    #pragma unroll
    for (int kb = 0; kb < 8; ++kb) {
      v16h b = *(const v16h*)(bp + kb * 32);
      c0 = __builtin_amdgcn_wmma_f32_16x16x32_f16(
               false, afrag[0][kb], false, b, (short)0, c0, false, false);
      c1 = __builtin_amdgcn_wmma_f32_16x16x32_f16(
               false, afrag[1][kb], false, b, (short)0, c1, false, false);
    }
    // Masked mining: cX[v] = sim[mBase+16t+v+8h][cBase+l16]
    #pragma unroll
    for (int v = 0; v < 8; ++v) {
      bool pos0 = (lr[0][v] == lc);
      bool pos1 = (lr[1][v] == lc);
      mn[0][v] = pos0 ? fminf(mn[0][v], c0[v]) : mn[0][v];
      mx[0][v] = pos0 ? mx[0][v] : fmaxf(mx[0][v], c0[v]);
      mn[1][v] = pos1 ? fminf(mn[1][v], c1[v]) : mn[1][v];
      mx[1][v] = pos1 ? mx[1][v] : fmaxf(mx[1][v], c1[v]);
    }
  }

  // Reduce across the 16 column lanes of each half (xor 1,2,4,8 stays in-half).
  #pragma unroll
  for (int t = 0; t < 2; ++t)
    #pragma unroll
    for (int v = 0; v < 8; ++v) {
      #pragma unroll
      for (int m = 8; m >= 1; m >>= 1) {
        mn[t][v] = fminf(mn[t][v], __shfl_xor(mn[t][v], m, 32));
        mx[t][v] = fmaxf(mx[t][v], __shfl_xor(mx[t][v], m, 32));
      }
    }

  __shared__ float sMin[8][MROWS], sMax[8][MROWS];
  if (l16 == 0) {
    #pragma unroll
    for (int t = 0; t < 2; ++t)
      #pragma unroll
      for (int v = 0; v < 8; ++v) {
        sMin[wave][t * 16 + 8 * h + v] = mn[t][v];
        sMax[wave][t * 16 + 8 * h + v] = mx[t][v];
      }
  }
  __syncthreads();

  __shared__ float sAp[MROWS], sAn[MROWS];
  if (tid < MROWS) {
    float ap = __builtin_inff(), an = -__builtin_inff();
    #pragma unroll
    for (int w = 0; w < 8; ++w) {
      ap = fminf(ap, sMin[w][tid]);
      an = fmaxf(an, sMax[w][tid]);
    }
    sAp[tid] = ap;  // dist_ap
    sAn[tid] = an;  // dist_an
  }
  __syncthreads();

  // d_p = ||x_row - ap + 1e-6||, d_n = ||x_row - an + 1e-6|| over D=256.
  // 8 threads per row, 32 elements each.
  const int   rr  = tid >> 3;           // 0..31
  const int   j   = tid & 7;
  const int   row = mBase + rr;
  const float ap  = sAp[rr], an = sAn[rr];
  const float* xr = x + (size_t)row * DD + j * 32;
  float sp2 = 0.f, sn2 = 0.f;
  #pragma unroll
  for (int d2 = 0; d2 < 32; ++d2) {
    const float xv = xr[d2];
    const float tp = xv - ap + 1e-6f;
    const float tn = xv - an + 1e-6f;
    sp2 += tp * tp;
    sn2 += tn * tn;
  }
  #pragma unroll
  for (int m = 4; m >= 1; m >>= 1) {    // rows occupy aligned 8-lane groups
    sp2 += __shfl_xor(sp2, m, 32);
    sn2 += __shfl_xor(sn2, m, 32);
  }
  if (j == 0)
    rowloss[row] = fmaxf(sqrtf(sp2) - sqrtf(sn2) + 1.0f, 0.0f);
}

// ---------- Phase 3: deterministic fixed-order scalar reduction ----------
__global__ __launch_bounds__(256) void k_sum(const float* __restrict__ rowloss,
                                             float* __restrict__ out) {
  const int tid = threadIdx.x;
  float s = 0.f;
  for (int i = tid; i < NN; i += 256) s += rowloss[i];
  __shared__ float sb[256];
  sb[tid] = s;
  __syncthreads();
  for (int step = 128; step >= 1; step >>= 1) {
    if (tid < step) sb[tid] += sb[tid + step];
    __syncthreads();
  }
  if (tid == 0) out[0] = sb[0];
}

extern "C" void kernel_launch(void* const* d_in, const int* in_sizes, int n_in,
                              void* d_out, int out_size, void* d_ws, size_t ws_size,
                              hipStream_t stream) {
  (void)in_sizes; (void)n_in; (void)out_size; (void)ws_size;
  const float* x      = (const float*)d_in[0];
  const int*   labels = (const int*)d_in[1];      // jax int32 labels
  _Float16* xn      = (_Float16*)d_ws;                                   // 4 MB
  float*    rowloss = (float*)((char*)d_ws + (size_t)NN * DD * sizeof(_Float16));

  k_normalize<<<NN, 256, 0, stream>>>(x, xn);
  k_mine<<<NN / MROWS, 256, 0, stream>>>(x, xn, labels, rowloss);
  k_sum<<<1, 256, 0, stream>>>(rowloss, (float*)d_out);
}
